// MultiHeadAttention_81544249081990
// MI455X (gfx1250) — matrix operand order repaired and checked by reading, not compiled, not run on hardware
//
#include <hip/hip_runtime.h>
#include <hip/hip_bf16.h>

#define HIDDEN 768
#define HEADS  12
#define DK     64
#define BSZ    4
#define SEQL   2048
#define MROWS  (BSZ * SEQL)   // 8192

typedef __attribute__((ext_vector_type(16))) _Float16 v16h;
typedef __attribute__((ext_vector_type(8)))  float    v8f;
typedef int v4i __attribute__((vector_size(16)));

#if __has_builtin(__builtin_amdgcn_global_load_async_to_lds_b128)
#define HAVE_ASYNC_LDS 1
#else
#define HAVE_ASYNC_LDS 0
#endif

typedef __attribute__((address_space(1))) v4i* gv4i_ptr;
typedef __attribute__((address_space(3))) v4i* lv4i_ptr;

union FragU { v16h v; uint4 q[2]; };

// Build a v16h WMMA fragment from two 16-byte chunks (A/B fragment halves).
__device__ __forceinline__ v16h load_frag2(const _Float16* p0, const _Float16* p1) {
    FragU u;
    u.q[0] = *(const uint4*)p0;
    u.q[1] = *(const uint4*)p1;
    return u.v;
}

__device__ __forceinline__ v8f zero_v8f() {
    v8f z;
#pragma unroll
    for (int i = 0; i < 8; ++i) z[i] = 0.0f;
    return z;
}

// 16-byte global -> LDS stage; async DMA path on gfx1250 when available.
__device__ __forceinline__ void stage16(_Float16* lds_dst, const _Float16* gsrc) {
#if HAVE_ASYNC_LDS
    __builtin_amdgcn_global_load_async_to_lds_b128(
        (gv4i_ptr)(gsrc),
        (lv4i_ptr)(lds_dst),
        0, 0);
#else
    *(uint4*)lds_dst = *(const uint4*)gsrc;
#endif
}

__device__ __forceinline__ void wait_stage() {
#if HAVE_ASYNC_LDS
#if __has_builtin(__builtin_amdgcn_s_wait_asynccnt)
    __builtin_amdgcn_s_wait_asynccnt(0);
#else
    asm volatile("s_wait_asynccnt 0x0" ::: "memory");
#endif
#endif
}

// ---------------------------------------------------------------------------
// Kernel 1: fp32 -> fp16 convert (grid-stride)
// ---------------------------------------------------------------------------
__global__ void cvt_f32_f16(const float* __restrict__ src,
                            _Float16* __restrict__ dst, int n) {
    int i = blockIdx.x * blockDim.x + threadIdx.x;
    int stride = gridDim.x * blockDim.x;
    for (; i < n; i += stride) dst[i] = (_Float16)src[i];
}

// ---------------------------------------------------------------------------
// Kernel 2/4: GEMM  Y = X @ W^T + bias   (M=8192, N=768, K=768)
// Block: 256 threads (8 waves). Block tile 256 (M) x 64 (N).
// Wave computes 32x64: two A row-tiles share every B fragment (halves LDS
// traffic per WMMA). W K-slab double-buffered in LDS, staged with async
// global->LDS DMA so the copy of slab t+1 overlaps WMMA on slab t.
//   mode 0: f16 out, [b,h,s,d] layout (QKV).  mode 1: f32 out, [m,e].
// ---------------------------------------------------------------------------
__global__ __launch_bounds__(256) void proj_gemm(
    const _Float16* __restrict__ X,
    const _Float16* __restrict__ W,
    const float*    __restrict__ bias,
    _Float16* __restrict__ out_h,
    float*    __restrict__ out_f,
    int mode)
{
    __shared__ __align__(16) _Float16 Ws[2][64 * 32];

    const int tid  = threadIdx.x;
    const int wave = tid >> 5;
    const int lane = tid & 31;
    const int lh   = lane >> 4;   // lane half (0/1)
    const int ln   = lane & 15;

    const int rb = blockIdx.x * 256;   // M base
    const int cb = blockIdx.y * 64;    // N base
    const int row0 = rb + wave * 32 + ln;
    const int row1 = row0 + 16;

    // staging role of this thread: 16B of the 64x32 W slab
    const int sr   = tid >> 2;         // 0..63  (W row within tile)
    const int sseg = tid & 3;          // 0..3   (16B segment within 32-half row)
    const _Float16* wsrc = &W[(size_t)(cb + sr) * HIDDEN + sseg * 8];

    v8f acc[2][4];
#pragma unroll
    for (int rt = 0; rt < 2; ++rt)
#pragma unroll
        for (int cc = 0; cc < 4; ++cc) acc[rt][cc] = zero_v8f();

    // prologue: stage slab 0
    stage16(&Ws[0][sr * 32 + sseg * 8], wsrc);
    wait_stage();
    __syncthreads();

    const int NT = HIDDEN / 32;  // 24 K-slabs
    for (int t = 0; t < NT; ++t) {
        const int p = t & 1;
        if (t + 1 < NT)
            stage16(&Ws[p ^ 1][sr * 32 + sseg * 8], wsrc + (t + 1) * 32);

        const int k0 = t * 32;
        const _Float16* ap0 = &X[(size_t)row0 * HIDDEN + k0];
        const _Float16* ap1 = &X[(size_t)row1 * HIDDEN + k0];
        v16h a0 = load_frag2(ap0 + lh * 8, ap0 + 16 + lh * 8);
        v16h a1 = load_frag2(ap1 + lh * 8, ap1 + 16 + lh * 8);

#pragma unroll
        for (int cc = 0; cc < 4; ++cc) {
            const _Float16* bp = &Ws[p][(cc * 16 + ln) * 32 + lh * 8];
            v16h b = load_frag2(bp, bp + 16);
            acc[0][cc] = __builtin_amdgcn_wmma_f32_16x16x32_f16(
                false, a0, false, b, (short)0, acc[0][cc], false, false);
            acc[1][cc] = __builtin_amdgcn_wmma_f32_16x16x32_f16(
                false, a1, false, b, (short)0, acc[1][cc], false, false);
        }

        wait_stage();
        __syncthreads();
    }

#pragma unroll
    for (int rt = 0; rt < 2; ++rt) {
#pragma unroll
        for (int cc = 0; cc < 4; ++cc) {
            const int col = cb + cc * 16 + ln;
            const float bv = bias[col];
#pragma unroll
            for (int r = 0; r < 8; ++r) {
                const int gm  = rb + wave * 32 + rt * 16 + r + 8 * lh;
                const float v = acc[rt][cc][r] + bv;
                if (mode == 0) {
                    const int bb = gm >> 11;        // / SEQL
                    const int s  = gm & (SEQL - 1);
                    const int h  = col >> 6;        // / DK
                    const int d  = col & (DK - 1);
                    out_h[(((size_t)bb * HEADS + h) * SEQL + s) * DK + d] = (_Float16)v;
                } else {
                    out_f[(size_t)gm * HIDDEN + col] = v;
                }
            }
        }
    }
}

// ---------------------------------------------------------------------------
// Kernel 3: flash attention over one (b, h), 128 queries per block.
// 8 waves x 16 queries each; key tiles of 32, double-buffered in LDS with
// async staging; online softmax in f32.
// ---------------------------------------------------------------------------
__global__ __launch_bounds__(256) void attn_fa(
    const _Float16* __restrict__ Q,   // [B,H,S,DK] f16
    const _Float16* __restrict__ K,
    const _Float16* __restrict__ V,
    _Float16* __restrict__ CTX)       // [B,S,HIDDEN] f16
{
    __shared__ __align__(16) _Float16 Kt[2][32 * 64];      // [key][d]
    __shared__ __align__(16) _Float16 Vt[2][64 * 32];      // [d][key] transposed
    __shared__ __align__(16) _Float16 Pb[8 * 16 * 32];     // per-wave P patch

    const int tid  = threadIdx.x;
    const int wave = tid >> 5;
    const int lane = tid & 31;
    const int lh   = lane >> 4;
    const int ln   = lane & 15;

    const int b = blockIdx.z, h = blockIdx.y;
    const int q0 = blockIdx.x * 128 + wave * 16;

    const size_t headOff = ((size_t)b * HEADS + h) * SEQL * DK;
    const _Float16* qp = Q + headOff;
    const _Float16* kp = K + headOff;
    const _Float16* vp = V + headOff;
    _Float16* Pw = &Pb[wave * 16 * 32];

    // V transpose staging role: this thread owns 8 contiguous elements of V
    const int vkey = (tid * 8) >> 6;        // key row (0..31)
    const int vd0  = (tid * 8) & 63;        // d base

    // Q A-fragments held in registers for the whole key loop
    const _Float16* qrow = qp + (size_t)(q0 + ln) * DK;
    v16h qa0 = load_frag2(qrow + lh * 16,      qrow + 8 + lh * 16);
    v16h qa1 = load_frag2(qrow + 32 + lh * 16, qrow + 40 + lh * 16);

    float m[8], l[8];
    v8f acc[4];
#pragma unroll
    for (int r = 0; r < 8; ++r) { m[r] = -1e30f; l[r] = 0.0f; }
#pragma unroll
    for (int cc = 0; cc < 4; ++cc) acc[cc] = zero_v8f();

    const float cscale = 0.18033688f;  // (1/sqrt(64)) * log2(e)

    // prologue: stage tile 0
    stage16(&Kt[0][tid * 8], kp + tid * 8);
#pragma unroll
    for (int j = 0; j < 8; ++j)
        Vt[0][(vd0 + j) * 32 + vkey] = vp[(size_t)vkey * DK + vd0 + j];
    wait_stage();
    __syncthreads();

    const int NT = SEQL / 32;  // 64 key tiles
    for (int t = 0; t < NT; ++t) {
        const int p = t & 1;
        if (t + 1 < NT) {
            const int kk1 = (t + 1) * 32;
            stage16(&Kt[p ^ 1][tid * 8], kp + (size_t)kk1 * DK + tid * 8);
#pragma unroll
            for (int j = 0; j < 8; ++j)
                Vt[p ^ 1][(vd0 + j) * 32 + vkey] =
                    vp[(size_t)(kk1 + vkey) * DK + vd0 + j];
        }

        // scores: S = Q @ K^T (two 16x16 tiles: keys 0-15, 16-31)
        v8f s0 = zero_v8f(), s1 = zero_v8f();
        {
            const _Float16* kb = &Kt[p][0];
            const _Float16* pp;
            pp = &kb[ln * 64 + lh * 16];
            v16h b00 = load_frag2(pp, pp + 8);
            pp = &kb[ln * 64 + 32 + lh * 16];
            v16h b01 = load_frag2(pp, pp + 8);
            s0 = __builtin_amdgcn_wmma_f32_16x16x32_f16(false, qa0, false, b00, (short)0, s0, false, false);
            s0 = __builtin_amdgcn_wmma_f32_16x16x32_f16(false, qa1, false, b01, (short)0, s0, false, false);
            pp = &kb[(16 + ln) * 64 + lh * 16];
            v16h b10 = load_frag2(pp, pp + 8);
            pp = &kb[(16 + ln) * 64 + 32 + lh * 16];
            v16h b11 = load_frag2(pp, pp + 8);
            s1 = __builtin_amdgcn_wmma_f32_16x16x32_f16(false, qa0, false, b10, (short)0, s1, false, false);
            s1 = __builtin_amdgcn_wmma_f32_16x16x32_f16(false, qa1, false, b11, (short)0, s1, false, false);
        }

        // online softmax (base-2); row r lives in a 16-lane half.
#pragma unroll
        for (int r = 0; r < 8; ++r) {
            float x0 = s0[r] * cscale, x1 = s1[r] * cscale;
            float mx = fmaxf(x0, x1);
#pragma unroll
            for (int off = 8; off > 0; off >>= 1)
                mx = fmaxf(mx, __shfl_xor(mx, off, 32));
            float mn    = fmaxf(m[r], mx);
            float alpha = exp2f(m[r] - mn);
            float p0 = exp2f(x0 - mn);
            float p1 = exp2f(x1 - mn);
            float rs = p0 + p1;
#pragma unroll
            for (int off = 8; off > 0; off >>= 1)
                rs += __shfl_xor(rs, off, 32);
            l[r] = l[r] * alpha + rs;
            m[r] = mn;
            int row = r + 8 * lh;
            Pw[row * 32 + ln]      = (_Float16)p0;
            Pw[row * 32 + 16 + ln] = (_Float16)p1;
#pragma unroll
            for (int cc = 0; cc < 4; ++cc) acc[cc][r] *= alpha;
        }

        // P as A fragment (16 q x 32 keys), per-wave private LDS patch
        v16h pa = load_frag2(&Pw[ln * 32 + lh * 16], &Pw[ln * 32 + 8 + lh * 16]);

        // ctx += P @ V  (4 d-chunks of 16)
#pragma unroll
        for (int cc = 0; cc < 4; ++cc) {
            const _Float16* pp = &Vt[p][(cc * 16 + ln) * 32 + lh * 16];
            v16h vb = load_frag2(pp, pp + 8);
            acc[cc] = __builtin_amdgcn_wmma_f32_16x16x32_f16(
                false, pa, false, vb, (short)0, acc[cc], false, false);
        }

        wait_stage();
        __syncthreads();
    }

    // normalize + write ctx merged as [b, s, h*64 + d] f16
#pragma unroll
    for (int r = 0; r < 8; ++r) {
        float inv = 1.0f / l[r];
        int q = q0 + r + 8 * lh;
#pragma unroll
        for (int cc = 0; cc < 4; ++cc) {
            int col = h * DK + cc * 16 + ln;
            CTX[((size_t)b * SEQL + q) * HIDDEN + col] = (_Float16)(acc[cc][r] * inv);
        }
    }
}

// ---------------------------------------------------------------------------
// Host-side launch
// ---------------------------------------------------------------------------
extern "C" void kernel_launch(void* const* d_in, const int* in_sizes, int n_in,
                              void* d_out, int out_size, void* d_ws, size_t ws_size,
                              hipStream_t stream) {
    (void)in_sizes; (void)n_in; (void)out_size; (void)ws_size;

    const float* x  = (const float*)d_in[0];
    const float* Wq = (const float*)d_in[1];
    const float* bq = (const float*)d_in[2];
    const float* Wk = (const float*)d_in[3];
    const float* bk = (const float*)d_in[4];
    const float* Wv = (const float*)d_in[5];
    const float* bv = (const float*)d_in[6];
    const float* Wo = (const float*)d_in[7];
    const float* bo = (const float*)d_in[8];
    float* out = (float*)d_out;

    const size_t NX = (size_t)MROWS * HIDDEN;     // 6,291,456
    const size_t NW = (size_t)HIDDEN * HIDDEN;    //   589,824

    _Float16* ws  = (_Float16*)d_ws;
    size_t off = 0;
    _Float16* xh   = ws + off; off += NX;
    _Float16* wqh  = ws + off; off += NW;
    _Float16* wkh  = ws + off; off += NW;
    _Float16* wvh  = ws + off; off += NW;
    _Float16* woh  = ws + off; off += NW;
    _Float16* qh   = ws + off; off += NX;   // [B,H,S,DK]
    _Float16* kh   = ws + off; off += NX;
    _Float16* vh   = ws + off; off += NX;
    _Float16* ctxh = ws + off; off += NX;   // [B,S,HIDDEN]

    // 1) convert inputs to f16
    cvt_f32_f16<<<2048, 256, 0, stream>>>(x,  xh,  (int)NX);
    cvt_f32_f16<<<512,  256, 0, stream>>>(Wq, wqh, (int)NW);
    cvt_f32_f16<<<512,  256, 0, stream>>>(Wk, wkh, (int)NW);
    cvt_f32_f16<<<512,  256, 0, stream>>>(Wv, wvh, (int)NW);
    cvt_f32_f16<<<512,  256, 0, stream>>>(Wo, woh, (int)NW);

    // 2) QKV projections (write head-split f16)
    dim3 ggrid(MROWS / 256, HIDDEN / 64);   // (32, 12)
    proj_gemm<<<ggrid, 256, 0, stream>>>(xh, wqh, bq, qh, nullptr, 0);
    proj_gemm<<<ggrid, 256, 0, stream>>>(xh, wkh, bk, kh, nullptr, 0);
    proj_gemm<<<ggrid, 256, 0, stream>>>(xh, wvh, bv, vh, nullptr, 0);

    // 3) flash attention
    attn_fa<<<dim3(SEQL / 128, HEADS, BSZ), 256, 0, stream>>>(qh, kh, vh, ctxh);

    // 4) output projection (fp32 result)
    proj_gemm<<<ggrid, 256, 0, stream>>>(ctxh, woh, bo, nullptr, out, 1);
}
